// DimeNetPlusPlus_24146306138666
// MI455X (gfx1250) — matrix operand order repaired and checked by reading, not compiled, not run on hardware
//
#include <hip/hip_runtime.h>
#include <hip/hip_bf16.h>
#include <cstdint>

// ---------------------------------------------------------------------------
// DimeNet++ forward for MI455X (gfx1250, wave32, WMMA + TDM).
// Dense matmuls -> v_wmma_f32_16x16x32_f16 (f16 in, f32 accum).
// Dominant GEMMs stage A-tiles into LDS with tensor_load_to_lds (TDM, padded
// rows for bank-conflict-free ds_load_b128 fragment reads).
// Irregular graph ops (gather / segment-sum) use VALU + global f32 atomics.
// ---------------------------------------------------------------------------

typedef __attribute__((ext_vector_type(16))) _Float16 v16h;
typedef __attribute__((ext_vector_type(8)))  _Float16 v8h;
typedef __attribute__((ext_vector_type(8)))  float    v8f;
typedef __attribute__((ext_vector_type(4)))  unsigned int u32x4;
typedef __attribute__((ext_vector_type(8)))  unsigned int u32x8;
typedef _Float16 h16;

namespace {
constexpr int Nn   = 20000;   // nodes
constexpr int Ee   = 200000;  // edges
constexpr int Tt   = 800000;  // triplets
constexpr int Hh   = 128;     // hidden
constexpr int INTC = 64;      // int_emb_size
constexpr int OUTC = 256;     // out_emb_channels
constexpr int NRr  = 6;       // num_radial
constexpr int SPH  = 42;      // NS*NR
constexpr int NBb  = 4;       // num_blocks
constexpr int Ll   = 3;       // output layers
constexpr float CUTOFF = 5.0f;
}

#define GF_SILU  1
#define GF_ACCUM 4

__device__ __forceinline__ float siluf(float x) { return x / (1.0f + __expf(-x)); }

union F16frag { v16h v; v8h h[2]; };

// ---------------------------------------------------------------------------
// Direct-global WMMA GEMM (fallback shapes): C[M,N] = ep(A[M,K] @ Bt[N,K]^T)
// Block = 8 waves; wave w -> rows [(blockIdx.x*8+w)*32, +32), cols blockIdx.y*64
// M%32==0, N%64==0, K%32==0.
// ---------------------------------------------------------------------------
__global__ __launch_bounds__(256)
void k_gemm(const h16* __restrict__ A, int lda,
            const h16* __restrict__ Bt, int ldb,
            const float* __restrict__ bias,
            const float* __restrict__ addsrc,
            float* __restrict__ outF, h16* __restrict__ outH,
            int M, int N, int K, int flags) {
  const int lane = threadIdx.x & 31;
  const int wave = threadIdx.x >> 5;
  const int row0 = (blockIdx.x * 8 + wave) * 32;
  if (row0 >= M) return;                       // wave-uniform: EXEC stays all-1s
  const int col0 = blockIdx.y * 64;
  const int lr = lane & 15;
  const int hi = (lane >> 4) & 1;

  v8f zf = {};
  v8f acc[2][4];
#pragma unroll
  for (int i = 0; i < 2; ++i)
#pragma unroll
    for (int j = 0; j < 4; ++j) acc[i][j] = zf;

  const h16* a0p = A + (size_t)(row0 + lr) * lda + hi * 8;
  const h16* a1p = a0p + (size_t)16 * lda;
  const h16* b0p = Bt + (size_t)(col0 + lr) * ldb + hi * 8;
  __builtin_prefetch(b0p, 0, 3);               // global_prefetch_b8: warm weights

  for (int k0 = 0; k0 < K; k0 += 32) {
    F16frag a0, a1, bf[4];
    a0.h[0] = *(const v8h*)(a0p + k0);
    a0.h[1] = *(const v8h*)(a0p + k0 + 16);
    a1.h[0] = *(const v8h*)(a1p + k0);
    a1.h[1] = *(const v8h*)(a1p + k0 + 16);
#pragma unroll
    for (int j = 0; j < 4; ++j) {
      const h16* bp = b0p + (size_t)(j * 16) * ldb + k0;
      bf[j].h[0] = *(const v8h*)(bp);
      bf[j].h[1] = *(const v8h*)(bp + 16);
    }
#pragma unroll
    for (int j = 0; j < 4; ++j) {
      acc[0][j] = __builtin_amdgcn_wmma_f32_16x16x32_f16(
          false, a0.v, false, bf[j].v, (short)0, acc[0][j], false, false);
      acc[1][j] = __builtin_amdgcn_wmma_f32_16x16x32_f16(
          false, a1.v, false, bf[j].v, (short)0, acc[1][j], false, false);
    }
  }

#pragma unroll
  for (int i2 = 0; i2 < 2; ++i2) {
#pragma unroll
    for (int j = 0; j < 4; ++j) {
#pragma unroll
      for (int f = 0; f < 8; ++f) {
        int m = row0 + i2 * 16 + hi * 8 + f;
        int n = col0 + j * 16 + lr;
        float r = acc[i2][j][f];
        if (bias) r += bias[n];
        if (flags & GF_SILU) r = siluf(r);
        size_t off = (size_t)m * N + n;
        if (addsrc) r += addsrc[off];
        if (outF) { if (flags & GF_ACCUM) outF[off] += r; else outF[off] = r; }
        if (outH) outH[off] = (h16)r;
      }
    }
  }
}

// ---------------------------------------------------------------------------
// TDM-staged WMMA GEMM for K in {64,128}, N%128==0, lda==K.
// Block tile = 128 rows x 128 cols; 8 waves as 4(row)x2(col), 32x64 each.
// Wave 0 DMAs the 128xK A-tile into LDS via tensor_load_to_lds with LDS
// padding of 4 DWORDs per 64 DWORDs => row stride 272B (K=128): lanes 0..15
// hit banks 0,4,...,60 -> conflict-free ds_load_b128 fragment reads.
// ---------------------------------------------------------------------------
template <int K>
__global__ __launch_bounds__(256)
void k_gemm_tdm(const h16* __restrict__ A, const h16* __restrict__ Bt,
                const float* __restrict__ bias,
                const float* __restrict__ addsrc,
                float* __restrict__ outF, h16* __restrict__ outH,
                int M, int N, int flags) {
  constexpr int TILE_BYTES = 128 * K * 2;
  constexpr int LDS_BYTES  = TILE_BYTES + (TILE_BYTES / 256) * 16;  // padded
  __shared__ __align__(16) unsigned char smem[LDS_BYTES];

  const int lane = threadIdx.x & 31;
  const int wave = threadIdx.x >> 5;
  const int rg   = wave & 3;                 // row group (0..3)
  const int cg   = wave >> 2;                // col group (0..1)
  const int rowB = blockIdx.x * 128;
  const int row0 = rowB + rg * 32;
  const int col0 = blockIdx.y * 128 + cg * 64;
  const int lr = lane & 15;
  const int hi = (lane >> 4) & 1;

  if (wave == 0) {
    unsigned long long gaddr =
        (unsigned long long)(uintptr_t)(A + (size_t)rowB * K);
    unsigned lds = (unsigned)(uintptr_t)(&smem[0]);   // low 32 bits = LDS offset
    u32x4 g0; u32x8 g1;
    u32x4 gz = {0u, 0u, 0u, 0u};
    // D# group 0: count=1 | lds_addr | global_addr[56:0] | type=2 ("image")
    g0[0] = 1u;
    g0[1] = lds;
    g0[2] = (unsigned)(gaddr & 0xFFFFFFFFu);
    g0[3] = (unsigned)((gaddr >> 32) & 0x01FFFFFFu) | (2u << 30);
    // D# group 1: data_size=2B | pad_enable | pad_interval=64dw | pad_amount=4dw
    g1[0] = (1u << 16) | (1u << 20) | (5u << 22) | (3u << 25);
    g1[1] = ((unsigned)K) << 16;                 // tensor_dim0[15:0] = K
    g1[2] = 0xFFFF0000u;                         // dim0 hi=0 | tensor_dim1 lo16
    g1[3] = 0x7FFFu | (((unsigned)K) << 16);     // tensor_dim1 hi | tile_dim0=K
    g1[4] = 128u;                                // tile_dim1=128, tile_dim2=0
    g1[5] = (unsigned)K;                         // tensor_dim0_stride lo32 = K
    g1[6] = 0u;
    g1[7] = 0u;
    asm volatile("tensor_load_to_lds %0, %1, %2, %3"
                 :: "s"(g0), "s"(g1), "s"(gz), "s"(gz)
                 : "memory");
    __builtin_amdgcn_s_wait_tensorcnt(0);
  }
  __syncthreads();

  v8f zf = {};
  v8f acc[2][4];
#pragma unroll
  for (int i = 0; i < 2; ++i)
#pragma unroll
    for (int j = 0; j < 4; ++j) acc[i][j] = zf;

  // padded LDS address of a data byte at linear offset L: L + (L/256)*16
  const unsigned lin0 = (unsigned)(rg * 32 + lr) * (K * 2);
  const unsigned lin1 = (unsigned)(rg * 32 + 16 + lr) * (K * 2);
  const unsigned char* s0 = smem + lin0 + (lin0 >> 8) * 16 + hi * 16;
  const unsigned char* s1 = smem + lin1 + (lin1 >> 8) * 16 + hi * 16;
  const h16* b0p = Bt + (size_t)(col0 + lr) * K + hi * 8;

#pragma unroll
  for (int k0 = 0; k0 < K; k0 += 32) {
    F16frag a0, a1, bf[4];
    a0.h[0] = *(const v8h*)(s0 + k0 * 2);
    a0.h[1] = *(const v8h*)(s0 + k0 * 2 + 32);
    a1.h[0] = *(const v8h*)(s1 + k0 * 2);
    a1.h[1] = *(const v8h*)(s1 + k0 * 2 + 32);
#pragma unroll
    for (int j = 0; j < 4; ++j) {
      const h16* bp = b0p + (size_t)(j * 16) * K + k0;
      bf[j].h[0] = *(const v8h*)(bp);
      bf[j].h[1] = *(const v8h*)(bp + 16);
    }
#pragma unroll
    for (int j = 0; j < 4; ++j) {
      acc[0][j] = __builtin_amdgcn_wmma_f32_16x16x32_f16(
          false, a0.v, false, bf[j].v, (short)0, acc[0][j], false, false);
      acc[1][j] = __builtin_amdgcn_wmma_f32_16x16x32_f16(
          false, a1.v, false, bf[j].v, (short)0, acc[1][j], false, false);
    }
  }

  if (row0 < M) {                              // wave-uniform store guard
#pragma unroll
    for (int i2 = 0; i2 < 2; ++i2) {
#pragma unroll
      for (int j = 0; j < 4; ++j) {
#pragma unroll
        for (int f = 0; f < 8; ++f) {
          int m = row0 + i2 * 16 + hi * 8 + f;
          int n = col0 + j * 16 + lr;
          float r = acc[i2][j][f];
          if (bias) r += bias[n];
          if (flags & GF_SILU) r = siluf(r);
          size_t off = (size_t)m * N + n;
          if (addsrc) r += addsrc[off];
          if (outF) { if (flags & GF_ACCUM) outF[off] += r; else outF[off] = r; }
          if (outH) outH[off] = (h16)r;
        }
      }
    }
  }
}

// ------------------------- small support kernels ---------------------------

__global__ void k_rbf(const float* __restrict__ dist, const float* __restrict__ freq,
                      float* __restrict__ rbf, int E) {
  int e = blockIdx.x * blockDim.x + threadIdx.x;
  if (e >= E) return;
  float d = dist[e] * (1.0f / CUTOFF);
  float d2 = d * d, d4 = d2 * d2, d5 = d4 * d;
  float env = 1.0f / d - 28.0f * d5 + 48.0f * d5 * d - 21.0f * d5 * d2;
  env *= (d < 1.0f) ? 1.0f : 0.0f;
#pragma unroll
  for (int r = 0; r < NRr; ++r)
    rbf[(size_t)e * NRr + r] = env * __sinf(freq[r] * d);
}

__global__ void k_sbfconv(const float* __restrict__ sbf, h16* __restrict__ out, int T) {
  int idx = blockIdx.x * blockDim.x + threadIdx.x;
  if (idx >= T * 64) return;
  int t = idx >> 6, c = idx & 63;
  out[idx] = (h16)((c < SPH) ? sbf[(size_t)t * SPH + c] : 0.0f);
}

__global__ void k_embA(const int* __restrict__ z, const int* __restrict__ gi,
                       const int* __restrict__ gj, const float* __restrict__ et,
                       const float* __restrict__ rbf, const float* __restrict__ Wr,
                       const float* __restrict__ br, h16* __restrict__ A, int E) {
  int idx = blockIdx.x * blockDim.x + threadIdx.x;
  if (idx >= E * 384) return;
  int e = idx / 384, c = idx % 384;
  float v;
  if (c < 128) {
    v = et[(size_t)z[gi[e]] * 128 + c];
  } else if (c < 256) {
    v = et[(size_t)z[gj[e]] * 128 + (c - 128)];
  } else {
    int hh = c - 256;
    float s = br[hh];
#pragma unroll
    for (int r = 0; r < NRr; ++r) s += rbf[(size_t)e * NRr + r] * Wr[r * 128 + hh];
    v = siluf(s);
  }
  A[idx] = (h16)v;
}

__global__ void k_wtrans(const float* __restrict__ W, h16* __restrict__ Wt,
                         int K, int N, int Kpad) {
  int idx = blockIdx.x * blockDim.x + threadIdx.x;
  if (idx >= N * Kpad) return;
  int n = idx / Kpad, k = idx % Kpad;
  Wt[idx] = (h16)((k < K) ? W[(size_t)k * N + n] : 0.0f);
}

__global__ void k_sbffuse(const float* __restrict__ W1, const float* __restrict__ W2,
                          h16* __restrict__ Wt) {
  int idx = blockIdx.x * blockDim.x + threadIdx.x;
  if (idx >= 64 * 64) return;
  int n = idx >> 6, k = idx & 63;
  float s = 0.0f;
  if (k < SPH) {
#pragma unroll
    for (int r = 0; r < 8; ++r) s += W1[k * 8 + r] * W2[r * 64 + n];
  }
  Wt[idx] = (h16)s;
}

__global__ void k_rbffuse(const float* __restrict__ W1, const float* __restrict__ W2,
                          float* __restrict__ Wf) {
  int idx = blockIdx.x * blockDim.x + threadIdx.x;
  if (idx >= NRr * Hh) return;
  int k = idx / Hh, n = idx % Hh;
  float s = 0.0f;
#pragma unroll
  for (int r = 0; r < 8; ++r) s += W1[k * 8 + r] * W2[r * Hh + n];
  Wf[idx] = s;
}

__global__ void k_mulrbfb(const float* __restrict__ xkj, const float* __restrict__ rbf,
                          const float* __restrict__ Wf, h16* __restrict__ out, int E) {
  int idx = blockIdx.x * blockDim.x + threadIdx.x;
  if (idx >= E * Hh) return;
  int e = idx >> 7, hh = idx & 127;
  float s = 0.0f;
#pragma unroll
  for (int r = 0; r < NRr; ++r) s += rbf[(size_t)e * NRr + r] * Wf[r * Hh + hh];
  out[idx] = (h16)(xkj[idx] * s);
}

__global__ void k_triplet(const h16* __restrict__ xkj, const h16* __restrict__ sbfb,
                          const int* __restrict__ kj, const int* __restrict__ ji,
                          float* __restrict__ acc, int T) {
  int idx = blockIdx.x * blockDim.x + threadIdx.x;
  if (idx >= T * 64) return;
  int t = idx >> 6, c = idx & 63;
  float v = (float)xkj[(size_t)kj[t] * 64 + c] * (float)sbfb[idx];
  atomicAdd(&acc[(size_t)ji[t] * 64 + c], v);
}

__global__ void k_f2h(const float* __restrict__ in, h16* __restrict__ out, int n) {
  int idx = blockIdx.x * blockDim.x + threadIdx.x;
  if (idx < n) out[idx] = (h16)in[idx];
}

__global__ void k_outedge(const float* __restrict__ rbf, const float* __restrict__ Wor,
                          const float* __restrict__ x, const int* __restrict__ gi,
                          float* __restrict__ nodeacc, int E) {
  int idx = blockIdx.x * blockDim.x + threadIdx.x;
  if (idx >= E * Hh) return;
  int e = idx >> 7, hh = idx & 127;
  float s = 0.0f;
#pragma unroll
  for (int r = 0; r < NRr; ++r) s += rbf[(size_t)e * NRr + r] * Wor[r * Hh + hh];
  atomicAdd(&nodeacc[(size_t)gi[e] * Hh + hh], s * x[idx]);
}

// --------------------------- host-side helpers -----------------------------

static inline void gemm(hipStream_t st, const h16* A, int lda, const h16* Bt, int ldb,
                        const float* bias, const float* addsrc,
                        float* outF, h16* outH, int M, int N, int K, int flags) {
  if ((K == 128 || K == 64) && (N % 128) == 0 && lda == K && ldb == K) {
    dim3 grid((M + 127) / 128, N / 128);
    if (K == 128)
      k_gemm_tdm<128><<<grid, 256, 0, st>>>(A, Bt, bias, addsrc, outF, outH, M, N, flags);
    else
      k_gemm_tdm<64><<<grid, 256, 0, st>>>(A, Bt, bias, addsrc, outF, outH, M, N, flags);
  } else {
    dim3 grid((M + 255) / 256, N / 64);
    k_gemm<<<grid, 256, 0, st>>>(A, lda, Bt, ldb, bias, addsrc, outF, outH, M, N, K, flags);
  }
}

static inline void wtrans(hipStream_t st, const float* W, h16* Wt, int K, int N, int Kpad) {
  int tot = N * Kpad;
  k_wtrans<<<(tot + 255) / 256, 256, 0, st>>>(W, Wt, K, N, Kpad);
}

extern "C" void kernel_launch(void* const* d_in, const int* in_sizes, int n_in,
                              void* d_out, int out_size, void* d_ws, size_t ws_size,
                              hipStream_t stream) {
  (void)in_sizes; (void)n_in; (void)out_size; (void)ws_size;
  const int*   z        = (const int*)  d_in[0];
  const float* dist     = (const float*)d_in[1];
  const float* sbf      = (const float*)d_in[2];
  const int*   gi       = (const int*)  d_in[3];
  const int*   gj       = (const int*)  d_in[4];
  const int*   idx_kj   = (const int*)  d_in[5];
  const int*   idx_ji   = (const int*)  d_in[6];
  const float* emb_table= (const float*)d_in[7];
  const float* W_rbf_emb= (const float*)d_in[8];
  const float* b_rbf_emb= (const float*)d_in[9];
  const float* W_emb    = (const float*)d_in[10];
  const float* b_emb    = (const float*)d_in[11];
  const float* freq     = (const float*)d_in[12];
  const float* Wi_rbf1  = (const float*)d_in[13];
  const float* Wi_rbf2  = (const float*)d_in[14];
  const float* Wi_sbf1  = (const float*)d_in[15];
  const float* Wi_sbf2  = (const float*)d_in[16];
  const float* Wi_kj    = (const float*)d_in[17];
  const float* bi_kj    = (const float*)d_in[18];
  const float* Wi_ji    = (const float*)d_in[19];
  const float* bi_ji    = (const float*)d_in[20];
  const float* Wi_down  = (const float*)d_in[21];
  const float* Wi_up    = (const float*)d_in[22];
  const float* Wi_rb_W1 = (const float*)d_in[23];
  const float* Wi_rb_b1 = (const float*)d_in[24];
  const float* Wi_rb_W2 = (const float*)d_in[25];
  const float* Wi_rb_b2 = (const float*)d_in[26];
  const float* Wi_lin   = (const float*)d_in[27];
  const float* bi_lin   = (const float*)d_in[28];
  const float* Wi_ra_W1 = (const float*)d_in[29];
  const float* Wi_ra_b1 = (const float*)d_in[30];
  const float* Wi_ra_W2 = (const float*)d_in[31];
  const float* Wi_ra_b2 = (const float*)d_in[32];
  const float* Wo_rbf   = (const float*)d_in[33];
  const float* Wo_up    = (const float*)d_in[34];
  const float* Wo_linW  = (const float*)d_in[35];
  const float* Wo_linb  = (const float*)d_in[36];

  // ---- workspace carve-up (deterministic) ----
  char* ws = (char*)d_ws;
  size_t cur = 0;
  auto alloc = [&](size_t bytes) -> char* {
    char* p = ws + cur;
    cur = (cur + bytes + 255) & ~(size_t)255;
    return p;
  };
  float* rbf     = (float*)alloc((size_t)Ee * NRr * 4);
  float* x32     = (float*)alloc((size_t)Ee * Hh * 4);     // edge features (f32 master)
  h16*   x16     = (h16*)  alloc((size_t)Ee * Hh * 2);     // f16 mirror (GEMM input)
  h16*   sbf16   = (h16*)  alloc((size_t)Tt * 64 * 2);     // padded sbf, persistent
  float* A1      = (float*)alloc((size_t)Ee * Hh * 4);     // x_ji
  float* A2      = (float*)alloc((size_t)Ee * Hh * 4);     // x_kj / hdd
  h16*   B1      = (h16*)  alloc((size_t)Ee * Hh * 2);     // f16 staging [E,H]
  h16*   T1      = (h16*)  alloc((size_t)Ee * Hh * 2);     // f16 staging [E,H]
  h16*   B2      = (h16*)  alloc((size_t)Ee * 64 * 2);     // x_kj down [E,64] f16
  float* acc     = (float*)alloc((size_t)Ee * 64 * 4);     // triplet segment-sum
  h16*   acc16   = (h16*)  alloc((size_t)Ee * 64 * 2);
  char*  R1      =         alloc((size_t)Ee * 384 * 2);    // Aemb16 / sbfb16 (aliased)
  h16*   Aemb16  = (h16*)R1;
  h16*   sbfb16  = (h16*)R1;
  float* nodeacc = (float*)alloc((size_t)Nn * Hh * 4);
  h16*   node16  = (h16*)  alloc((size_t)Nn * Hh * 2);
  h16*   C1      = (h16*)  alloc((size_t)Nn * OUTC * 2);
  h16*   C2      = (h16*)  alloc((size_t)Nn * OUTC * 2);
  h16*   wt0     = (h16*)  alloc((size_t)256 * 384 * 2);   // transposed-weight scratch
  h16*   wsbf    = (h16*)  alloc((size_t)64 * 64 * 2);
  float* wrbf    = (float*)alloc((size_t)NRr * Hh * 4);
  float* outP    = (float*)d_out;                           // [N, 256]

  auto out_block = [&](int b) {
    (void)hipMemsetAsync(nodeacc, 0, (size_t)Nn * Hh * 4, stream);
    k_outedge<<<(Ee * Hh + 255) / 256, 256, 0, stream>>>(
        rbf, Wo_rbf + (size_t)b * NRr * Hh, x32, gi, nodeacc, Ee);
    k_f2h<<<(Nn * Hh + 255) / 256, 256, 0, stream>>>(nodeacc, node16, Nn * Hh);
    wtrans(stream, Wo_up + (size_t)b * Hh * OUTC, wt0, Hh, OUTC, Hh);
    gemm(stream, node16, Hh, wt0, Hh, nullptr, nullptr, nullptr, C1, Nn, OUTC, Hh, 0);
    for (int l = 0; l < Ll; ++l) {
      const float* Wl = Wo_linW + ((size_t)b * Ll + l) * OUTC * OUTC;
      const float* bl = Wo_linb + ((size_t)b * Ll + l) * OUTC;
      wtrans(stream, Wl, wt0, OUTC, OUTC, OUTC);
      if (l == 0)
        gemm(stream, C1, OUTC, wt0, OUTC, bl, nullptr, nullptr, C2, Nn, OUTC, OUTC, GF_SILU);
      else if (l == 1)
        gemm(stream, C2, OUTC, wt0, OUTC, bl, nullptr, nullptr, C1, Nn, OUTC, OUTC, GF_SILU);
      else  // last layer: P += silu(...)
        gemm(stream, C1, OUTC, wt0, OUTC, bl, nullptr, outP, nullptr, Nn, OUTC, OUTC,
             GF_SILU | GF_ACCUM);
    }
  };

  // ---- front end ----
  (void)hipMemsetAsync(outP, 0, (size_t)Nn * OUTC * 4, stream);
  k_rbf<<<(Ee + 255) / 256, 256, 0, stream>>>(dist, freq, rbf, Ee);
  k_sbfconv<<<(Tt * 64 + 255) / 256, 256, 0, stream>>>(sbf, sbf16, Tt);
  k_embA<<<(Ee * 384 + 255) / 256, 256, 0, stream>>>(
      z, gi, gj, emb_table, rbf, W_rbf_emb, b_rbf_emb, Aemb16, Ee);
  wtrans(stream, W_emb, wt0, 384, Hh, 384);
  gemm(stream, Aemb16, 384, wt0, 384, b_emb, nullptr, x32, x16, Ee, Hh, 384, GF_SILU);
  out_block(0);

  // ---- interaction blocks ----
  for (int b = 0; b < NBb; ++b) {
    // x_ji = silu(x @ Wji + b)
    wtrans(stream, Wi_ji + (size_t)b * Hh * Hh, wt0, Hh, Hh, Hh);
    gemm(stream, x16, Hh, wt0, Hh, bi_ji + (size_t)b * Hh, nullptr, A1, nullptr,
         Ee, Hh, Hh, GF_SILU);
    // x_kj = silu(x @ Wkj + b)
    wtrans(stream, Wi_kj + (size_t)b * Hh * Hh, wt0, Hh, Hh, Hh);
    gemm(stream, x16, Hh, wt0, Hh, bi_kj + (size_t)b * Hh, nullptr, A2, nullptr,
         Ee, Hh, Hh, GF_SILU);
    // x_kj *= rbf @ (Wrbf1@Wrbf2)   -> f16
    k_rbffuse<<<(NRr * Hh + 255) / 256, 256, 0, stream>>>(
        Wi_rbf1 + (size_t)b * NRr * 8, Wi_rbf2 + (size_t)b * 8 * Hh, wrbf);
    k_mulrbfb<<<(Ee * Hh + 255) / 256, 256, 0, stream>>>(A2, rbf, wrbf, B1, Ee);
    // down-project: silu(. @ Wdown) -> [E,64] f16
    wtrans(stream, Wi_down + (size_t)b * Hh * INTC, wt0, Hh, INTC, Hh);
    gemm(stream, B1, Hh, wt0, Hh, nullptr, nullptr, nullptr, B2, Ee, INTC, Hh, GF_SILU);
    // sbf_b = sbf @ (Wsbf1@Wsbf2)  [T,64]
    k_sbffuse<<<(64 * 64 + 255) / 256, 256, 0, stream>>>(
        Wi_sbf1 + (size_t)b * SPH * 8, Wi_sbf2 + (size_t)b * 8 * INTC, wsbf);
    gemm(stream, sbf16, 64, wsbf, 64, nullptr, nullptr, nullptr, sbfb16, Tt, INTC, 64, 0);
    // triplet gather * sbf_b, segment-sum over idx_ji
    (void)hipMemsetAsync(acc, 0, (size_t)Ee * INTC * 4, stream);
    k_triplet<<<(Tt * 64 + 255) / 256, 256, 0, stream>>>(B2, sbfb16, idx_kj, idx_ji, acc, Tt);
    k_f2h<<<(Ee * INTC + 255) / 256, 256, 0, stream>>>(acc, acc16, Ee * INTC);
    // up-project + add x_ji: hdd = x_ji + silu(acc @ Wup)
    wtrans(stream, Wi_up + (size_t)b * INTC * Hh, wt0, INTC, Hh, INTC);
    gemm(stream, acc16, INTC, wt0, INTC, nullptr, A1, A2, B1, Ee, Hh, INTC, GF_SILU);
    // residual before skip: hdd += silu(silu(hdd@W1+b1)@W2+b2)
    wtrans(stream, Wi_rb_W1 + (size_t)b * Hh * Hh, wt0, Hh, Hh, Hh);
    gemm(stream, B1, Hh, wt0, Hh, Wi_rb_b1 + (size_t)b * Hh, nullptr, nullptr, T1,
         Ee, Hh, Hh, GF_SILU);
    wtrans(stream, Wi_rb_W2 + (size_t)b * Hh * Hh, wt0, Hh, Hh, Hh);
    gemm(stream, T1, Hh, wt0, Hh, Wi_rb_b2 + (size_t)b * Hh, A2, A2, B1,
         Ee, Hh, Hh, GF_SILU);
    // skip: x = silu(hdd@Wlin+b) + x
    wtrans(stream, Wi_lin + (size_t)b * Hh * Hh, wt0, Hh, Hh, Hh);
    gemm(stream, B1, Hh, wt0, Hh, bi_lin + (size_t)b * Hh, x32, x32, x16,
         Ee, Hh, Hh, GF_SILU);
    // residual after skip
    wtrans(stream, Wi_ra_W1 + (size_t)b * Hh * Hh, wt0, Hh, Hh, Hh);
    gemm(stream, x16, Hh, wt0, Hh, Wi_ra_b1 + (size_t)b * Hh, nullptr, nullptr, T1,
         Ee, Hh, Hh, GF_SILU);
    wtrans(stream, Wi_ra_W2 + (size_t)b * Hh * Hh, wt0, Hh, Hh, Hh);
    gemm(stream, T1, Hh, wt0, Hh, Wi_ra_b2 + (size_t)b * Hh, x32, x32, x16,
         Ee, Hh, Hh, GF_SILU);
    out_block(b + 1);
  }
}
